// PEDS_85830626443513
// MI455X (gfx1250) — compile-verified, matrix operand order checked by reading, not run on hardware
//
#include <hip/hip_runtime.h>
#include <hip/hip_bf16.h>

typedef __attribute__((ext_vector_type(16))) _Float16 v16h;
typedef __attribute__((ext_vector_type(8)))  float    v8f;
typedef __attribute__((ext_vector_type(4)))  float    f4;
typedef __attribute__((ext_vector_type(4)))  unsigned u32x4;
typedef __attribute__((ext_vector_type(4)))  int      i32x4;
typedef __attribute__((ext_vector_type(8)))  int      i32x8;

#define RESN   100
#define NCELLS (RESN * RESN)   // 10000
#define BATCH  32

// Jacobi LDS layout: 102 halo rows x PITCH floats, interior col j at GOFF+j.
// PITCH multiple of 4 floats and GOFF multiple of 4 -> all span bases 16B-aligned.
#define PITCH  112
#define GOFF   4
#define NROWH  102

// Tensor Data Mover availability (probe-verified builtin on both toolchains,
// but with different arity: ROCm7.2/clang-22 = 5 args, clang-23 = 6 args).
#if defined(__has_builtin)
#if __has_builtin(__builtin_amdgcn_tensor_load_to_lds) && \
    __has_builtin(__builtin_amdgcn_s_wait_tensorcnt)
#define PEDS_HAVE_TDM 1
#endif
#endif

// ---------------------------------------------------------------------------
// Layer 1: h1 = relu(pores(32x25) @ W1(25x128) + b1)   grid=32, block=128
// ---------------------------------------------------------------------------
__global__ __launch_bounds__(128) void peds_layer1(
    const float* __restrict__ pores, const float* __restrict__ W1,
    const float* __restrict__ b1, float* __restrict__ h1)
{
    __shared__ float x[25];
    const int m = blockIdx.x, n = threadIdx.x;
    if (n < 25) x[n] = pores[m * 25 + n];
    __syncthreads();
    float acc = b1[n];
#pragma unroll
    for (int k = 0; k < 25; ++k) acc = fmaf(x[k], W1[k * 128 + n], acc);
    h1[m * 128 + n] = fmaxf(acc, 0.0f);
}

// ---------------------------------------------------------------------------
// Layer 2: h2 = relu(h1(32x128) @ W2(128x256) + b2)    grid=32, block=256
// ---------------------------------------------------------------------------
__global__ __launch_bounds__(256) void peds_layer2(
    const float* __restrict__ h1, const float* __restrict__ W2,
    const float* __restrict__ b2, float* __restrict__ h2)
{
    __shared__ float x[128];
    const int m = blockIdx.x, n = threadIdx.x;
    if (n < 128) x[n] = h1[m * 128 + n];
    __syncthreads();
    float acc = b2[n];
#pragma unroll 8
    for (int k = 0; k < 128; ++k) acc = fmaf(x[k], W2[k * 256 + n], acc);
    h2[m * 256 + n] = fmaxf(acc, 0.0f);
}

// ---------------------------------------------------------------------------
// Layer 3: h3 = relu(h2(32x256) @ W3(256x512) + b3), stored as f16 (WMMA A)
// grid=32, block=512
// ---------------------------------------------------------------------------
__global__ __launch_bounds__(512) void peds_layer3(
    const float* __restrict__ h2, const float* __restrict__ W3,
    const float* __restrict__ b3, _Float16* __restrict__ h3h)
{
    __shared__ float x[256];
    const int m = blockIdx.x, n = threadIdx.x;
    if (n < 256) x[n] = h2[m * 256 + n];
    __syncthreads();
    float acc = b3[n];
#pragma unroll 8
    for (int k = 0; k < 256; ++k) acc = fmaf(x[k], W3[k * 512 + n], acc);
    h3h[m * 512 + n] = (_Float16)fmaxf(acc, 0.0f);
}

// ---------------------------------------------------------------------------
// Layer 4 via WMMA: gen = h3(32x512) @ W4(512x10000) + b4,
// cond = max(gen + (1 - pore_map_upsampled), 0.01), written to d_out cond slab.
// One wave per 16-wide N tile, full M=32 (two 16x16 C tiles), K swept by 32.
// HBM-bound on the single 20.5MB W4 read; f16 convert in registers, f32 accum.
// grid = ceil(625/8) = 79 blocks, block = 256 (8 waves).
// ---------------------------------------------------------------------------
__global__ __launch_bounds__(256) void peds_gemm4_wmma(
    const _Float16* __restrict__ A,      // h3 f16, 32x512 row-major
    const float* __restrict__ W4,        // 512x10000 row-major
    const float* __restrict__ b4,
    const float* __restrict__ pores,
    float* __restrict__ cond)            // 32x10000
{
    const int wave = threadIdx.x >> 5;
    const int lane = threadIdx.x & 31;
    const int tile = blockIdx.x * 8 + wave;
    if (tile >= NCELLS / 16) return;     // 625 N tiles
    const int n0   = tile * 16;
    const int lhi  = lane >> 4;          // 0: lanes 0-15, 1: lanes 16-31
    const int lmod = lane & 15;

    v8f c0 = {};
    v8f c1 = {};

#pragma unroll 2
    for (int k0 = 0; k0 < 512; k0 += 32) {
        if (k0 + 64 < 512)               // stream W4 two k-blocks ahead
            __builtin_prefetch(&W4[(k0 + 64) * NCELLS + n0 + lmod], 0, 1);
        v16h a0, a1, b;
#pragma unroll
        for (int i = 0; i < 16; ++i) {
            const int kk = k0 + ((i < 8) ? i : i + 8) + (lhi << 3);
            a0[i] = A[lmod * 512 + kk];                 // M tile 0 (rows 0-15)
            a1[i] = A[(lmod + 16) * 512 + kk];          // M tile 1 (rows 16-31)
            b[i]  = (_Float16)W4[kk * NCELLS + n0 + lmod];
        }
        c0 = __builtin_amdgcn_wmma_f32_16x16x32_f16(
                 false, a0, false, b, (short)0, c0, false, false);
        c1 = __builtin_amdgcn_wmma_f32_16x16x32_f16(
                 false, a1, false, b, (short)0, c1, false, false);
    }

    // Fused epilogue: + bias + residual base conductivity, clamp to 0.01
    const int n    = n0 + lmod;          // flat index into 100x100 grid
    const float bias = b4[n];
    const int gi = n / RESN, gj = n % RESN;
    const int pcol = (gi / 20) * 5 + (gj / 20);
#pragma unroll
    for (int r = 0; r < 8; ++r) {
        const int m0 = r + (lhi << 3);           // C/D layout: M = r + 8*laneHi
        const float v0 = c0[r] + bias + (1.0f - pores[m0 * 25 + pcol]);
        cond[m0 * NCELLS + n] = fmaxf(v0, 0.01f);
        const int m1 = m0 + 16;
        const float v1 = c1[r] + bias + (1.0f - pores[m1 * 25 + pcol]);
        cond[m1 * NCELLS + n] = fmaxf(v1, 0.01f);
    }
}

// ---------------------------------------------------------------------------
// LDS-resident Jacobi relaxation, 1000 sweeps. One workgroup per sample.
// Setup: the 40KB cond tile is DMA'd into LDS by the Tensor Data Mover
// (one TENSOR_LOAD_TO_LDS per block, overlapped with T-buffer init, waited
// on TENSORcnt), replacing ~50K scalar global gathers.
// Sweeps: each active thread owns a 20-cell horizontal span; all LDS traffic
// is 16B-aligned float4 (ds_load_b128/ds_store_b128). Double-buffered halo
// grids + staging = ~129KB of the 320KB/WGP LDS. Top ghost row = 1
// (Dirichlet), bottom ghost = 0, side ghosts refreshed each sweep (Neumann).
// grid = 32, block = 512 (16 waves, 500 active span threads).
// ---------------------------------------------------------------------------
__global__ __launch_bounds__(512) void peds_jacobi(
    const float* __restrict__ cond, float* __restrict__ kappa)
{
    __shared__ __align__(16) float Ta[NROWH * PITCH];
    __shared__ __align__(16) float Tb[NROWH * PITCH];
    __shared__ __align__(16) float kbuf[NCELLS];   // staged cond tile
    __shared__ float red[128];

    const int m   = blockIdx.x;
    const int tid = threadIdx.x;
    const float* __restrict__ kg = cond + m * NCELLS;

#ifdef PEDS_HAVE_TDM
    // --- TDM: DMA the contiguous 100x100 f32 cond tile into kbuf ----------
    if (tid < 32) {                      // wave 0 only (wave-uniform branch)
        const unsigned long long ga = (unsigned long long)(uintptr_t)kg;
        // AMDGPU flat->LDS mapping keeps addr[31:0] as the LDS byte offset.
        const unsigned lds_off = (unsigned)(uintptr_t)kbuf;
        u32x4 g0;
        g0[0] = 1u;                                          // count=1 (valid D#)
        g0[1] = lds_off;                                     // D#.lds_addr
        g0[2] = (unsigned)(ga & 0xFFFFFFFFu);                // global_addr[31:0]
        g0[3] = (unsigned)((ga >> 32) & 0x1FFFFFFu)          // global_addr[56:32]
              | (2u << 30);                                  // type = 2 ("image")
        i32x8 g1;
        g1[0] = (int)(2u << 16);                             // data_size=4B, no mask
        g1[1] = (int)((NCELLS & 0xFFFF) << 16);              // tensor_dim0 lo16
        g1[2] = (int)((NCELLS >> 16) | (1u << 16));          // dim0 hi | tensor_dim1=1
        g1[3] = (int)((NCELLS & 0xFFFF) << 16);              // tile_dim0 = 10000
        g1[4] = 1;                                           // tile_dim1 = 1
        g1[5] = NCELLS;                                      // tensor_dim0_stride
        g1[6] = 0; g1[7] = 0;
        const i32x4 gz = {0, 0, 0, 0};                       // 2D: groups 2/3 unused
#if __clang_major__ >= 23
        const i32x8 gz8 = {0, 0, 0, 0, 0, 0, 0, 0};
        __builtin_amdgcn_tensor_load_to_lds(g0, g1, gz, gz, gz8, 0);
#else
        __builtin_amdgcn_tensor_load_to_lds(g0, g1, gz, gz, 0);
#endif
    }
#else
    for (int idx = tid; idx < NCELLS; idx += 512) kbuf[idx] = kg[idx];
#endif

    // init: T(row i) = 1 - i/99 on interior rows, ghost rows 1.0 / 0.0
    // (overlaps with the in-flight tensor DMA)
    for (int idx = tid; idx < NROWH * PITCH; idx += 512) {
        const int i = idx / PITCH;
        float v;
        if (i == 0)              v = 1.0f;
        else if (i >= NROWH - 1) v = 0.0f;
        else                     v = 1.0f - (float)(i - 1) * (1.0f / 99.0f);
        Ta[idx] = v;
        Tb[idx] = v;
    }

#ifdef PEDS_HAVE_TDM
    if (tid < 32) __builtin_amdgcn_s_wait_tensorcnt(0);
#endif
    __syncthreads();                     // kbuf + T buffers visible to all

    const bool active = tid < 500;               // 100 rows x 5 spans
    const int  r    = tid / 5;                   // grid row 0..99
    const int  j0   = (tid % 5) * 20;            // span start col (16B-aligned)
    const int  rowN = r * PITCH + GOFF + j0;     // north halo row
    const int  rowC = (r + 1) * PITCH + GOFF;    // own halo row (col base)
    const int  rowS = (r + 2) * PITCH + GOFF + j0;

    // per-cell stencil coefficients (face conductivity / denom), BCs folded
    float cN[20], cS[20], cW[20], cE[20];
    if (active) {
#pragma unroll
        for (int jj = 0; jj < 20; ++jj) {
            const int j = j0 + jj;
            const int cell = r * RESN + j;
            const float kc = kbuf[cell];
            const float kn = (r == 0)        ? kc : 0.5f * (kc + kbuf[cell - RESN]);
            const float ks = (r == RESN - 1) ? kc : 0.5f * (kc + kbuf[cell + RESN]);
            const float kw = (j == 0)        ? kc : 0.5f * (kc + kbuf[cell - 1]);
            const float ke = (j == RESN - 1) ? kc : 0.5f * (kc + kbuf[cell + 1]);
            const float inv = 1.0f / (kn + ks + kw + ke + 1e-12f);
            cN[jj] = kn * inv; cS[jj] = ks * inv;
            cW[jj] = kw * inv; cE[jj] = ke * inv;
        }
    } else {
#pragma unroll
        for (int jj = 0; jj < 20; ++jj) cN[jj] = cS[jj] = cW[jj] = cE[jj] = 0.0f;
    }
    __syncthreads();

    float* cur = Ta;
    float* nxt = Tb;
    for (int it = 0; it < 1000; ++it) {
        if (active) {
            f4 nv[5], sv[5], cv[7], ov[5];
#pragma unroll
            for (int q = 0; q < 5; ++q) nv[q] = *(const f4*)&cur[rowN + 4 * q];
#pragma unroll
            for (int q = 0; q < 5; ++q) sv[q] = *(const f4*)&cur[rowS + 4 * q];
#pragma unroll
            for (int q = 0; q < 7; ++q) cv[q] = *(const f4*)&cur[rowC + j0 - 4 + 4 * q];
#pragma unroll
            for (int jj = 0; jj < 20; ++jj) {
                const float tn = nv[jj >> 2][jj & 3];
                const float ts = sv[jj >> 2][jj & 3];
                const float tw = cv[(jj + 3) >> 2][(jj + 3) & 3];
                const float te = cv[(jj + 5) >> 2][(jj + 5) & 3];
                float v = cN[jj] * tn;
                v = fmaf(cS[jj], ts, v);
                v = fmaf(cW[jj], tw, v);
                v = fmaf(cE[jj], te, v);
                ov[jj >> 2][jj & 3] = v;
            }
#pragma unroll
            for (int q = 0; q < 5; ++q) *(f4*)&nxt[rowC + j0 + 4 * q] = ov[q];
            if (j0 == 0)  nxt[rowC - 1]   = ov[0][0];   // west Neumann ghost
            if (j0 == 80) nxt[rowC + 100] = ov[4][3];   // east Neumann ghost
        }
        __syncthreads();
        float* t = cur; cur = nxt; nxt = t;
    }

    // kappa = mean_j 2*res*k(0,j)*(1 - T(0,j)) = 2 * sum_j k(0,j)*(1-T(0,j))
    if (tid < RESN) red[tid] = kbuf[tid] * (1.0f - cur[PITCH + GOFF + tid]);
    __syncthreads();
    if (tid == 0) {
        float s = 0.0f;
        for (int j = 0; j < RESN; ++j) s += red[j];
        kappa[m] = 2.0f * s;
    }
}

// ---------------------------------------------------------------------------
extern "C" void kernel_launch(void* const* d_in, const int* in_sizes, int n_in,
                              void* d_out, int out_size, void* d_ws, size_t ws_size,
                              hipStream_t stream)
{
    const float* pores = (const float*)d_in[0];
    const float* W1    = (const float*)d_in[1];
    const float* b1    = (const float*)d_in[2];
    const float* W2    = (const float*)d_in[3];
    const float* b2    = (const float*)d_in[4];
    const float* W3    = (const float*)d_in[5];
    const float* b3    = (const float*)d_in[6];
    const float* W4    = (const float*)d_in[7];
    const float* b4    = (const float*)d_in[8];

    float* out   = (float*)d_out;
    float* kappa = out;                  // (32,)
    float* cond  = out + BATCH;          // (32,100,100)

    float*    h1  = (float*)d_ws;                   // 32x128 f32
    float*    h2  = h1 + BATCH * 128;               // 32x256 f32
    _Float16* h3h = (_Float16*)(h2 + BATCH * 256);  // 32x512 f16

    peds_layer1<<<BATCH, 128, 0, stream>>>(pores, W1, b1, h1);
    peds_layer2<<<BATCH, 256, 0, stream>>>(h1, W2, b2, h2);
    peds_layer3<<<BATCH, 512, 0, stream>>>(h2, W3, b3, h3h);
    peds_gemm4_wmma<<<(NCELLS / 16 + 7) / 8, 256, 0, stream>>>(h3h, W4, b4, pores, cond);
    peds_jacobi<<<BATCH, 512, 0, stream>>>(cond, kappa);
}
